// AFA_35270271434855
// MI455X (gfx1250) — matrix operand order repaired and checked by reading, not compiled, not run on hardware
//
#include <hip/hip_runtime.h>

typedef __attribute__((ext_vector_type(16))) _Float16 v16h;
typedef __attribute__((ext_vector_type(8)))  _Float16 v8h;
typedef __attribute__((ext_vector_type(4)))  _Float16 h4;
typedef __attribute__((ext_vector_type(2)))  _Float16 h2;
typedef __attribute__((ext_vector_type(8)))  float    v8f;

// ---------------------------------------------------------------------------
// WMMA GEMM:  C(M,N) fp32 = A(M,K) x B(K,N), fp32 operands converted to f16
// while staging tiles through LDS.  B-operand producers (compile-time MODE):
//   0: dense fp32 (K,N) row-major
//   1: implicit 3x3 SAME conv, K = 9*Cin over SRC laid out (Cin, B*HW)
//   2: fused align-corners bilinear resize of SRC (Bn, Cin, Hi, Wi)
// GUARD_K / GUARD_N: branchless clamp+select bounds handling, only
// instantiated for the launches that need them (K%32!=0 or N%64!=0).
// Tiles: 128 thr (4 waves) -> 16(M) x 64(N), K step 32.
// LDS layouts chosen so fragment gathers are 2x ds_load_b128 per operand:
//   As[row][k]  row-major 16x32 halves
//   Bs[n][k]    n-major   64x32 halves
// ---------------------------------------------------------------------------
struct GemmP {
  const float* A;
  const float* B;
  const float* SRC;
  float*       C;
  int M, N, K;
  int H, W, HW;      // stage output spatial (modes 1/2 n-decode)
  int Cin;
  int Hi, Wi;        // resize source spatial
  float sy, sx;      // resize scales (align corners)
  int Bn;
};

__device__ __forceinline__ float conv_sample(const GemmP& P, int kk, int n) {
  int ch = kk / 9, p = kk - ch * 9;
  int dy = p / 3 - 1, dx = p - (p / 3) * 3 - 1;
  int b = n / P.HW;  int hw = n - b * P.HW;
  int hr = hw / P.W;
  int h = hr + dy;
  int w = hw - hr * P.W + dx;
  bool ok = (h >= 0) & (h < P.H) & (w >= 0) & (w < P.W);
  int hc = ok ? h : 0, wc = ok ? w : 0;
  float v = P.SRC[(size_t)ch * P.N + b * P.HW + hc * P.W + wc];
  return ok ? v : 0.f;
}

__device__ __forceinline__ float bilin_sample(const GemmP& P, int cch, int b,
                                              int y0, int y1, int x0, int x1,
                                              float wy, float wx) {
  const float* s = P.SRC + ((size_t)b * P.Cin + cch) * (size_t)(P.Hi * P.Wi);
  float v00 = s[y0 * P.Wi + x0], v01 = s[y0 * P.Wi + x1];
  float v10 = s[y1 * P.Wi + x0], v11 = s[y1 * P.Wi + x1];
  return v00 * (1 - wy) * (1 - wx) + v01 * (1 - wy) * wx
       + v10 * wy * (1 - wx)       + v11 * wy * wx;
}

template <int MODE, bool GK, bool GN>
__global__ __launch_bounds__(128) void gemm_wmma_kernel(GemmP P) {
  __shared__ alignas(16) _Float16 As[16 * 32];
  __shared__ alignas(16) _Float16 Bs[64 * 32];
  const int tid  = threadIdx.x;
  const int wv   = tid >> 5;
  const int ln   = tid & 31;
  const int half = ln >> 4;
  const int l16  = ln & 15;
  const int m0   = blockIdx.y * 16;
  const int n0   = blockIdx.x * 64;
  v8f acc = {};
  for (int kb = 0; kb < P.K; kb += 32) {
    // ---- stage A tile 16x32: one float4 load + one b64 LDS store ----
    if (!GK) {
      int row = tid >> 3, col4 = (tid & 7) * 4;
      const float4 a4 =
          *(const float4*)(P.A + (size_t)(m0 + row) * P.K + kb + col4);
      h4 h = {(_Float16)a4.x, (_Float16)a4.y, (_Float16)a4.z, (_Float16)a4.w};
      *(h4*)(As + row * 32 + col4) = h;
    } else {
      for (int i = tid; i < 512; i += 128) {
        int r = i >> 5, c = i & 31;
        int kk = kb + c;
        int kc = kk < P.K ? kk : P.K - 1;           // clamped addr
        float v = P.A[(size_t)(m0 + r) * P.K + kc]; // unconditional load
        As[i] = (_Float16)(kk < P.K ? v : 0.f);     // select
      }
    }
    // ---- stage B tile 32x64 into n-major Bs[n][k], packed f16 pairs ----
#pragma unroll
    for (int i = 0; i < 8; i++) {
      int idx = i * 128 + tid;          // 0..1023 : (k-pair, n)
      int kp  = idx >> 6;               // 0..15
      int c   = idx & 63;
      int k0  = kb + kp * 2;
      int n   = n0 + c;
      float v0, v1;
      if (MODE == 0) {
        int nc  = GN ? (n < P.N ? n : P.N - 1) : n;
        int ka0 = GK ? (k0 < P.K ? k0 : P.K - 1) : k0;
        int ka1 = GK ? (k0 + 1 < P.K ? k0 + 1 : P.K - 1) : k0 + 1;
        v0 = P.B[(size_t)ka0 * P.N + nc];
        v1 = P.B[(size_t)ka1 * P.N + nc];
        if (GK) { v0 = (k0     < P.K) ? v0 : 0.f;
                  v1 = (k0 + 1 < P.K) ? v1 : 0.f; }
        if (GN) { if (n >= P.N) { v0 = 0.f; v1 = 0.f; } }
      } else if (MODE == 1) {
        v0 = conv_sample(P, k0, n);
        v1 = conv_sample(P, k0 + 1, n);
      } else {
        int b  = n / P.HW;  int hw = n - b * P.HW;
        int ho = hw / P.W;  int wo = hw - ho * P.W;
        float yf = ho * P.sy, xf = wo * P.sx;
        int y0 = (int)yf, x0 = (int)xf;
        int y1 = (y0 + 1 < P.Hi) ? y0 + 1 : P.Hi - 1;
        int x1 = (x0 + 1 < P.Wi) ? x0 + 1 : P.Wi - 1;
        float wy = yf - (float)y0, wx = xf - (float)x0;
        v0 = bilin_sample(P, k0,     b, y0, y1, x0, x1, wy, wx);
        v1 = bilin_sample(P, k0 + 1, b, y0, y1, x0, x1, wy, wx);
      }
      h2 pr = {(_Float16)v0, (_Float16)v1};
      *(h2*)(Bs + c * 32 + kp * 2) = pr;
    }
    __syncthreads();
    // ---- fragment gather: 2x b128 per operand (k = 16*half + s for both A and B) ----
    const v8h* Ap = (const v8h*)(As + l16 * 32 + 16 * half);
    const v8h* Bp = (const v8h*)(Bs + (wv * 16 + l16) * 32 + 16 * half);
    v8h a0 = Ap[0], a1 = Ap[1];
    v8h b0 = Bp[0], b1 = Bp[1];
    v16h af = __builtin_shufflevector(a0, a1, 0, 1, 2, 3, 4, 5, 6, 7, 8, 9, 10,
                                      11, 12, 13, 14, 15);
    v16h bf = __builtin_shufflevector(b0, b1, 0, 1, 2, 3, 4, 5, 6, 7, 8, 9, 10,
                                      11, 12, 13, 14, 15);
    acc = __builtin_amdgcn_wmma_f32_16x16x32_f16(false, af, false, bf,
                                                 (short)0, acc, false, false);
    __syncthreads();
  }
  // ---- store D: VGPR r -> row m0 + r + 8*half, col = lane%16 ----
  int ncol = n0 + wv * 16 + l16;
  if (!GN || ncol < P.N) {
#pragma unroll
    for (int r = 0; r < 8; r++)
      P.C[(size_t)(m0 + r + 8 * half) * P.N + ncol] = acc[r];
  }
}

// ---------------------------------------------------------------------------
// GroupNorm in-place over x laid out (C, N), N = Bn*HW.
// One block per (batch, group); fused stats + apply; optional ReLU.
// ---------------------------------------------------------------------------
__global__ __launch_bounds__(256) void gn_kernel(float* x, const float* gamma,
                                                 const float* beta, int C, int G,
                                                 int HW, int N, int doRelu) {
  int b = blockIdx.x / G, g = blockIdx.x % G;
  int cpg = C / G;
  int c0 = g * cpg, col0 = b * HW;
  int tot = cpg * HW;
  float s = 0.f, s2 = 0.f;
  for (int i = threadIdx.x; i < tot; i += 256) {
    int c = c0 + i / HW, col = col0 + i % HW;
    float v = x[(size_t)c * N + col];
    s += v; s2 += v * v;
  }
  __shared__ float rs[256], rq[256];
  rs[threadIdx.x] = s; rq[threadIdx.x] = s2;
  __syncthreads();
  for (int o = 128; o > 0; o >>= 1) {
    if (threadIdx.x < o) { rs[threadIdx.x] += rs[threadIdx.x + o];
                           rq[threadIdx.x] += rq[threadIdx.x + o]; }
    __syncthreads();
  }
  __shared__ float smu, srstd;
  if (threadIdx.x == 0) {
    float m = rs[0] / (float)tot;
    float var = rq[0] / (float)tot - m * m;
    smu = m; srstd = rsqrtf(var + 1e-5f);
  }
  __syncthreads();
  float mu = smu, rstd = srstd;
  for (int i = threadIdx.x; i < tot; i += 256) {
    int c = c0 + i / HW, col = col0 + i % HW;
    size_t ix = (size_t)c * N + col;
    float v = (x[ix] - mu) * rstd * gamma[c] + beta[c];
    if (doRelu) v = fmaxf(v, 0.f);
    x[ix] = v;
  }
}

__global__ void sum4_kernel(float* out, const float* f0, const float* f1,
                            const float* f2, const float* f3, size_t total) {
  size_t i = (size_t)blockIdx.x * blockDim.x + threadIdx.x;
  if (i < total) out[i] = f0[i] + f1[i] + f2[i] + f3[i];
}

// s[b*C + c] = mean over HW of x[c][b*HW + :]
__global__ __launch_bounds__(256) void rowmean_kernel(const float* x, float* s,
                                                      int C, int HW, int N) {
  int b = blockIdx.x / C, c = blockIdx.x % C;
  float acc = 0.f;
  for (int i = threadIdx.x; i < HW; i += 256) acc += x[(size_t)c * N + b * HW + i];
  __shared__ float r[256];
  r[threadIdx.x] = acc; __syncthreads();
  for (int o = 128; o > 0; o >>= 1) {
    if (threadIdx.x < o) r[threadIdx.x] += r[threadIdx.x + o];
    __syncthreads();
  }
  if (threadIdx.x == 0) s[b * C + c] = r[0] / (float)HW;
}

// relu(s @ fc1^T) @ fc2^T -> softmax -> top-2 gated weights (B,4)
__global__ __launch_bounds__(256) void selector_kernel(const float* s,
                                                       const float* fc1,
                                                       const float* fc2,
                                                       float* wsel, int chs,
                                                       int Bn) {
  __shared__ float hid[2560];   // Bn * chs/4 <= 32*80
  __shared__ float zsh[128];
  int h4v = chs / 4;
  for (int idx = threadIdx.x; idx < Bn * h4v; idx += 256) {
    int b = idx / h4v, j = idx - b * h4v;
    float acc = 0.f;
    for (int c = 0; c < chs; c++) acc += s[b * chs + c] * fc1[j * chs + c];
    hid[idx] = fmaxf(acc, 0.f);
  }
  __syncthreads();
  for (int idx = threadIdx.x; idx < Bn * 4; idx += 256) {
    int b = idx >> 2, o = idx & 3;
    float acc = 0.f;
    for (int j = 0; j < h4v; j++) acc += hid[b * h4v + j] * fc2[o * h4v + j];
    zsh[idx] = acc;
  }
  __syncthreads();
  if ((int)threadIdx.x < Bn) {
    int b = threadIdx.x;
    float z0 = zsh[b*4], z1 = zsh[b*4+1], z2 = zsh[b*4+2], z3 = zsh[b*4+3];
    float zm = fmaxf(fmaxf(z0, z1), fmaxf(z2, z3));
    float e0 = expf(z0-zm), e1 = expf(z1-zm), e2 = expf(z2-zm), e3 = expf(z3-zm);
    float es = e0 + e1 + e2 + e3;
    float sel[4] = {e0/es, e1/es, e2/es, e3/es};
    int i1 = 0;
    for (int j = 1; j < 4; j++) if (sel[j] > sel[i1]) i1 = j;          // first max
    int i2 = -1;
    for (int j = 0; j < 4; j++) if (j != i1 && (i2 < 0 || sel[j] > sel[i2])) i2 = j;
    for (int j = 0; j < 4; j++) wsel[b*4+j] = (j == i1 || j == i2) ? sel[j] : 0.f;
  }
}

// v = relu( sum_i wsel[b][i] * feas_i )  over (C,N)
__global__ void selsum_kernel(float* v, const float* f0, const float* f1,
                              const float* f2, const float* f3,
                              const float* wsel, int HW, int N, size_t total) {
  size_t i = (size_t)blockIdx.x * blockDim.x + threadIdx.x;
  if (i >= total) return;
  int n = (int)(i % (size_t)N);
  int b = n / HW;
  float r = wsel[b*4+0]*f0[i] + wsel[b*4+1]*f1[i]
          + wsel[b*4+2]*f2[i] + wsel[b*4+3]*f3[i];
  v[i] = fmaxf(r, 0.f);
}

// per-column channel LN stats over x (C,N)
__global__ void colstats_kernel(const float* x, float* mu, float* rstd,
                                int C, int N) {
  int n = blockIdx.x * blockDim.x + threadIdx.x;
  if (n >= N) return;
  float s = 0.f, s2 = 0.f;
  for (int c = 0; c < C; c++) { float v = x[(size_t)c * N + n]; s += v; s2 += v*v; }
  float m = s / (float)C;
  float var = s2 / (float)C - m * m;
  mu[n] = m; rstd[n] = rsqrtf(var + 1e-5f);
}

// pcat[(off+c)*Bn + b] = w[c] * mean_hw((x - mu)*rstd) + bias[c]
__global__ __launch_bounds__(256) void lnpool_kernel(const float* x,
                                                     const float* mu,
                                                     const float* rstd,
                                                     const float* w,
                                                     const float* bvec,
                                                     float* pcat, int C, int HW,
                                                     int N, int off, int Bn) {
  int b = blockIdx.x / C, c = blockIdx.x % C;
  float acc = 0.f;
  for (int i = threadIdx.x; i < HW; i += 256) {
    int n = b * HW + i;
    acc += (x[(size_t)c * N + n] - mu[n]) * rstd[n];
  }
  __shared__ float r[256];
  r[threadIdx.x] = acc; __syncthreads();
  for (int o = 128; o > 0; o >>= 1) {
    if (threadIdx.x < o) r[threadIdx.x] += r[threadIdx.x + o];
    __syncthreads();
  }
  if (threadIdx.x == 0)
    pcat[(size_t)(off + c) * Bn + b] = w[c] * (r[0] / (float)HW) + bvec[c];
}

// out[b*512 + o] = outT[o*Bn + b] + bias[o]
__global__ void bias_tr_kernel(const float* outT, const float* bias, float* out,
                               int Bn) {
  int idx = blockIdx.x * blockDim.x + threadIdx.x;
  if (idx >= 512 * Bn) return;
  int b = idx / 512, o = idx - b * 512;
  out[idx] = outT[(size_t)o * Bn + b] + bias[o];
}

// ---------------------------------------------------------------------------
template <int MODE, bool GK, bool GNF>
static inline void run_gemm(const GemmP& P, hipStream_t s) {
  dim3 grid((P.N + 63) / 64, P.M / 16);
  gemm_wmma_kernel<MODE, GK, GNF><<<grid, dim3(128), 0, s>>>(P);
}

extern "C" void kernel_launch(void* const* d_in, const int* in_sizes, int n_in,
                              void* d_out, int out_size, void* d_ws,
                              size_t ws_size, hipStream_t stream) {
  (void)in_sizes; (void)n_in; (void)out_size; (void)ws_size;
  const int Bn = 32;
  const int CHS[3]  = {64, 128, 320};
  const int HS[3]   = {56, 28, 14};
  const int CIN[4]  = {64, 128, 320, 512};
  const int HIN[4]  = {56, 28, 14, 7};
  const int stageOff[3] = {0, 64, 192};

  const float* X[4];
  for (int i = 0; i < 4; i++) X[i] = (const float*)d_in[i];
  auto pp = [&](int i) -> const float* { return (const float*)d_in[4 + i]; };
  // params flat index map (insertion-order DFS):
  //   msfs[k] at 34*k: merge[i]={w,g,b}@3i ; branch[m]@12+11m =
  //     {w1,g1,b1,w2,g2,b2,fc1,fc2,cw,cg,cb}
  //   ln_p[k] {w,b} @ 102+2k ; ln_q[k] @ 108+2k ; red1_w 114 red1_b 115
  //   red2_w 116 red2_b 117

  float* W = (float*)d_ws;
  const size_t CN = (size_t)64 * 100352;  // max chs*N (stage 0)
  float* feas[4] = {W, W + CN, W + 2 * CN, W + 3 * CN};
  float* fsum = W + 4 * CN;
  float* u1   = W + 5 * CN;
  float* u2   = W + 6 * CN;
  float* vbuf = W + 7 * CN;
  float* cbuf = W + 8 * CN;
  float* colmu = W + 9 * CN;
  float* colrs = colmu + 100352;
  float* svec  = colrs + 100352;
  float* wsel  = svec + Bn * 320;
  float* pcat[2];
  pcat[0] = wsel + Bn * 4;
  pcat[1] = pcat[0] + 512 * Bn;
  float* outT = pcat[1] + 512 * Bn;

  for (int k = 0; k < 3; k++) {
    const int chs = CHS[k], Hh = HS[k], HW = Hh * Hh, N = Bn * HW;
    const int G    = (chs / 4 < 32) ? chs / 4 : 32;
    const int chs4 = chs / 4;
    const int G1   = (chs4 / 4 < 32) ? chs4 / 4 : 32;
    const int mb   = 34 * k;
    const size_t total = (size_t)chs * (size_t)N;

    // ---- merge: fused resize + 1x1 conv (WMMA) + GN, for each input ----
    for (int i = 0; i < 4; i++) {
      GemmP P = {};
      P.A = pp(mb + 3 * i); P.SRC = X[i]; P.C = feas[i];
      P.M = chs; P.N = N; P.K = CIN[i];
      P.H = Hh; P.W = Hh; P.HW = HW;
      P.Cin = CIN[i]; P.Hi = HIN[i]; P.Wi = HIN[i];
      P.sy = (float)(HIN[i] - 1) / (float)(Hh - 1); P.sx = P.sy; P.Bn = Bn;
      run_gemm<2, false, false>(P, stream);
      gn_kernel<<<Bn * G, 256, 0, stream>>>(feas[i], pp(mb + 3 * i + 1),
                                            pp(mb + 3 * i + 2), chs, G, HW, N, 0);
    }
    sum4_kernel<<<(int)((total + 255) / 256), 256, 0, stream>>>(
        fsum, feas[0], feas[1], feas[2], feas[3], total);

    for (int m = 0; m < 2; m++) {
      const int bb = mb + 12 + 11 * m;
      GemmP P1 = {}; P1.A = pp(bb + 0); P1.B = fsum; P1.C = u1;
      P1.M = chs4; P1.N = N; P1.K = chs;
      run_gemm<0, false, false>(P1, stream);
      gn_kernel<<<Bn * G1, 256, 0, stream>>>(u1, pp(bb + 1), pp(bb + 2),
                                             chs4, G1, HW, N, 1);
      GemmP P2 = {}; P2.A = pp(bb + 3); P2.B = u1; P2.C = u2;
      P2.M = chs; P2.N = N; P2.K = chs4;
      if (chs4 % 32 == 0) run_gemm<0, false, false>(P2, stream);
      else                run_gemm<0, true,  false>(P2, stream);
      gn_kernel<<<Bn * G, 256, 0, stream>>>(u2, pp(bb + 4), pp(bb + 5),
                                            chs, G, HW, N, 0);
      rowmean_kernel<<<Bn * chs, 256, 0, stream>>>(u2, svec, chs, HW, N);
      selector_kernel<<<1, 256, 0, stream>>>(svec, pp(bb + 6), pp(bb + 7),
                                             wsel, chs, Bn);
      selsum_kernel<<<(int)((total + 255) / 256), 256, 0, stream>>>(
          vbuf, feas[0], feas[1], feas[2], feas[3], wsel, HW, N, total);
      // 3x3 conv as implicit GEMM (K = 9*chs) on WMMA
      GemmP P3 = {}; P3.A = pp(bb + 8); P3.SRC = vbuf; P3.C = cbuf;
      P3.M = chs; P3.N = N; P3.K = 9 * chs;
      P3.H = Hh; P3.W = Hh; P3.HW = HW; P3.Cin = chs;
      run_gemm<1, false, false>(P3, stream);
      gn_kernel<<<Bn * G, 256, 0, stream>>>(cbuf, pp(bb + 9), pp(bb + 10),
                                            chs, G, HW, N, 1);
      colstats_kernel<<<(N + 255) / 256, 256, 0, stream>>>(cbuf, colmu, colrs,
                                                           chs, N);
      const float* lnw = pp((m == 0 ? 102 : 108) + 2 * k);
      const float* lnb = pp((m == 0 ? 102 : 108) + 2 * k + 1);
      lnpool_kernel<<<Bn * chs, 256, 0, stream>>>(cbuf, colmu, colrs, lnw, lnb,
                                                  pcat[m], chs, HW, N,
                                                  stageOff[k], Bn);
    }
  }

  // ---- final reductions: outT(512,32) = red_w(512,512) @ pcat(512,32) ----
  for (int m = 0; m < 2; m++) {
    GemmP PF = {}; PF.A = pp(114 + 2 * m); PF.B = pcat[m]; PF.C = outT;
    PF.M = 512; PF.N = Bn; PF.K = 512;
    run_gemm<0, false, true>(PF, stream);
    float* outp = (float*)d_out + (size_t)m * 512 * Bn;
    bias_tr_kernel<<<(512 * Bn + 255) / 256, 256, 0, stream>>>(
        outT, pp(115 + 2 * m), outp, Bn);
  }
}
